// MessagePassingConvolution_66314295050827
// MI455X (gfx1250) — matrix operand order repaired and checked by reading, not compiled, not run on hardware
//
#include <hip/hip_runtime.h>

// ---------------------------------------------------------------------------
// MessagePassingConvolution for gfx1250 (MI455X).
//   * 4-layer edge MLP (8->64->64->64->256) as f16 WMMA / f32 accum,
//     16 edges per wave32.
//   * Weights staged once per block into LDS, pre-swizzled into WMMA
//     B-fragment order -> each B operand is a single contiguous v16h
//     (2x ds_load_b128) per lane.
//   * Activations staged through LDS in A-fragment-swizzled order.
//   * Gather / tensor products / scatter-add with global f32 atomics
//     (node_feats 20.5MB and out 41MB are L2-resident on a 192MB-L2 part).
// ---------------------------------------------------------------------------

#define NODE_DIM   256   // MUL + 3*MUL
#define EDGE_DIM   11    // NUM_RADIAL + 3
#define OUT_DIM    512   // 2*MUL + 2*MUL*3

typedef __attribute__((ext_vector_type(16))) _Float16 v16h;
typedef __attribute__((ext_vector_type(8)))  float    v8f;

__device__ __forceinline__ float silu_act(float x) {
    // silu(x) * ACT_CST with fast v_rcp_f32 (no IEEE divide expansion).
    const float ACT_CST = 1.6791767f;
    return x * __builtin_amdgcn_rcpf(1.0f + __expf(-x)) * ACT_CST;
}

// ---------------------------------------------------------------------------
// LDS layout (dynamic):
//   [0, 65536)  float mixbuf[4 waves][16][256]
//   then f16 region (fragment-swizzled):
//     abufS[4 waves][2 kc][32 lanes][16 halves]   (4096 halves)
//     w0S  [4 t][32][16]                          (2048 halves, K>=8 zero-pad)
//     w1S  [4 t * 2 kc][32][16]                   (4096 halves)
//     w2S  [4 t * 2 kc][32][16]                   (4096 halves)
//     w3S  [16 t * 2 kc][32][16]                  (16384 halves)
// Fragment conventions (best-effort per CDNA5 ISA 7.12.2, wave32):
//   A 16x32 f16: lane L holds row m=L&15, group g=L>>4;
//                half (2j+lo) = element k = 2j + 8g + 8*(j>=4) + lo.
//   B 32x16 f16: lane L holds col n=L&15, group g=L>>4;
//                half (2j+lo) = element k = 2j + 8g + 8*(j>=4) + lo.
//   D 16x16 f32: lane L, vgpr r = element (m = r + 8g, n = L&15).
// ---------------------------------------------------------------------------
#define MIX_BYTES    (4 * 16 * 256 * 4)
#define ABUFS_HALVES (4 * 2 * 512)
#define W0S_HALVES   (4 * 512)
#define W1S_HALVES   (8 * 512)
#define W2S_HALVES   (8 * 512)
#define W3S_HALVES   (32 * 512)
#define LDS_BYTES    (MIX_BYTES + 2 * (ABUFS_HALVES + W0S_HALVES + W1S_HALVES + W2S_HALVES + W3S_HALVES))

// Stage a KxN f32 weight matrix into LDS in B-fragment order.
// NT = N/16 n-tiles, KC = k-chunks (32 rows each); rows k >= K are zero-padded.
__device__ __forceinline__ void stage_weight(const float* __restrict__ W,
                                             _Float16* __restrict__ dst,
                                             int NT, int KC, int K, int N,
                                             int tid, int nthreads)
{
    const int total8 = (NT * KC * 512) / 8;          // groups of 8 halves
    for (int d8 = tid; d8 < total8; d8 += nthreads) {
        const int d    = d8 * 8;
        const int tile = d >> 9;                     // tile = t*KC + kc
        const int rem  = d & 511;
        const int lane = rem >> 4;
        const int h0   = rem & 15;                   // 0 or 8
        const int g  = lane >> 4;
        const int lm = lane & 15;
        const int kc = tile % KC;
        const int t  = tile / KC;
        const int n  = 16 * t + lm;
        union { _Float16 h[8]; uint4 u; } pk;
        #pragma unroll
        for (int q = 0; q < 8; ++q) {
            const int hh = h0 + q;
            const int j  = hh >> 1;
            const int lo = hh & 1;
            const int k  = kc * 32 + 8 * g + (hh & 7) + 16 * (hh >> 3);
            pk.h[q] = (k < K) ? (_Float16)W[k * N + n] : (_Float16)0.0f;
        }
        *(uint4*)(dst + d) = pk.u;                   // ds_store_b128
    }
}

// Scatter one f32 activation (row m, feature n) into the A-fragment buffer.
__device__ __forceinline__ void store_act(_Float16* __restrict__ abufS,
                                          int m, int n, float h)
{
    const int kc = n >> 5;
    const int k  = n & 31;
    const int lo = k & 1;
    const int gg = (k >> 3) & 1;
    const int j  = ((k >> 1) & 3) + 4 * (k >> 4);
    abufS[kc * 512 + (16 * gg + m) * 16 + 2 * j + lo] = (_Float16)h;
}

__global__ void mp_conv_kernel(const float* __restrict__ node_feats,
                               const float* __restrict__ edge_attrs,
                               const int*   __restrict__ senders,
                               const int*   __restrict__ receivers,
                               const float* __restrict__ w0,
                               const float* __restrict__ w1,
                               const float* __restrict__ w2,
                               const float* __restrict__ w3,
                               float* __restrict__ out,
                               int E)
{
    extern __shared__ char smem[];
    float*    mix_all   = (float*)smem;
    _Float16* abufS_all = (_Float16*)(smem + MIX_BYTES);
    _Float16* w0S = abufS_all + ABUFS_HALVES;
    _Float16* w1S = w0S + W0S_HALVES;
    _Float16* w2S = w1S + W1S_HALVES;
    _Float16* w3S = w2S + W2S_HALVES;

    const int tid = threadIdx.x;
    const int nth = (int)blockDim.x;

    // ---- stage weights (f32 -> f16, fragment-swizzled, b128 stores) ----
    stage_weight(w0, w0S, 4, 1, 8, 64, tid, nth);
    stage_weight(w1, w1S, 4, 2, 64, 64, tid, nth);
    stage_weight(w2, w2S, 4, 2, 64, 64, tid, nth);
    stage_weight(w3, w3S, 16, 2, 64, 256, tid, nth);
    __syncthreads();

    const int wave = tid >> 5;
    const int lane = tid & 31;
    const int lm   = lane & 15;
    const int g    = lane >> 4;

    float*    mixbuf = mix_all   + wave * (16 * 256);
    _Float16* abufS  = abufS_all + wave * (2 * 512);

    const int  tile  = blockIdx.x * 4 + wave;
    const long eBase = (long)tile * 16;

    // Prefetch this tile's sender rows (global_prefetch_b8) for the edge phase.
    if (g == 0) {
        long e = eBase + lm;
        if (e < (long)E) {
            int snd = senders[e];
            __builtin_prefetch(node_feats + (size_t)snd * NODE_DIM, 0, 1);
        }
    }

    // =======================================================================
    // Layer 0:  es(16x8, K padded to 32) @ w0(8x64), scale 1/sqrt(8), SiLU
    // =======================================================================
    v16h a0 = {};
    {
        long e = eBase + lm;
        if (g == 0 && e < (long)E) {                 // g=1 holds K=8..15 -> 0
            const float* esp = edge_attrs + (size_t)e * EDGE_DIM;
            #pragma unroll
            for (int j = 0; j < 4; ++j) {
                a0[2 * j]     = (_Float16)esp[2 * j];
                a0[2 * j + 1] = (_Float16)esp[2 * j + 1];
            }
        }
    }
    {
        const float inv_sqrt8 = 0.35355339059327373f;
        #pragma unroll
        for (int t = 0; t < 4; ++t) {
            v16h b = *(const v16h*)(w0S + t * 512 + lane * 16);  // 2x ds_load_b128
            v8f c = {};
            c = __builtin_amdgcn_wmma_f32_16x16x32_f16(false, a0, false, b,
                                                       (short)0, c, false, false);
            #pragma unroll
            for (int r = 0; r < 8; ++r)
                store_act(abufS, r + 8 * g, 16 * t + lm,
                          silu_act(c[r] * inv_sqrt8));
        }
    }
    __syncthreads();

    // =======================================================================
    // Layers 1 & 2:  h(16x64) @ w(64x64), scale 1/8, SiLU
    // =======================================================================
    _Float16* wh[2] = { w1S, w2S };
    #pragma unroll
    for (int layer = 0; layer < 2; ++layer) {
        _Float16* W = wh[layer];
        const v16h a_lo = *(const v16h*)(abufS +       lane * 16);
        const v16h a_hi = *(const v16h*)(abufS + 512 + lane * 16);
        v8f acc[4];
        #pragma unroll
        for (int t = 0; t < 4; ++t) {
            v8f c = {};
            v16h b0 = *(const v16h*)(W + (t * 2 + 0) * 512 + lane * 16);
            c = __builtin_amdgcn_wmma_f32_16x16x32_f16(false, a_lo, false, b0,
                                                       (short)0, c, false, false);
            v16h b1 = *(const v16h*)(W + (t * 2 + 1) * 512 + lane * 16);
            c = __builtin_amdgcn_wmma_f32_16x16x32_f16(false, a_hi, false, b1,
                                                       (short)0, c, false, false);
            acc[t] = c;
        }
        __syncthreads();                             // all reads of abufS done
        #pragma unroll
        for (int t = 0; t < 4; ++t)
            #pragma unroll
            for (int r = 0; r < 8; ++r)
                store_act(abufS, r + 8 * g, 16 * t + lm,
                          silu_act(acc[t][r] * 0.125f));
        __syncthreads();
    }

    // =======================================================================
    // Layer 3:  h(16x64) @ w3(64x256), scale 1/8 -> mixbuf (f32)
    // =======================================================================
    {
        const v16h a_lo = *(const v16h*)(abufS +       lane * 16);
        const v16h a_hi = *(const v16h*)(abufS + 512 + lane * 16);
        #pragma unroll
        for (int t = 0; t < 16; ++t) {
            v8f c = {};
            v16h b0 = *(const v16h*)(w3S + (t * 2 + 0) * 512 + lane * 16);
            c = __builtin_amdgcn_wmma_f32_16x16x32_f16(false, a_lo, false, b0,
                                                       (short)0, c, false, false);
            v16h b1 = *(const v16h*)(w3S + (t * 2 + 1) * 512 + lane * 16);
            c = __builtin_amdgcn_wmma_f32_16x16x32_f16(false, a_hi, false, b1,
                                                       (short)0, c, false, false);
            #pragma unroll
            for (int r = 0; r < 8; ++r)
                mixbuf[(r + 8 * g) * 256 + 16 * t + lm] = c[r] * 0.125f;
        }
    }
    __syncthreads();

    // =======================================================================
    // Edge phase: gather sender features, form messages, scatter-add.
    //   out per node: [ms*mix | tp_s*mix | mv*mix (64x3) | tp_v*mix (64x3)]
    // =======================================================================
    const float OUT_SCALE = 0.25f;                   // 1/sqrt(AVG_NEIGH)
    const float INV_SQRT3 = 0.57735026918962576f;
    for (int m = 0; m < 16; ++m) {
        const long e = eBase + m;
        if (e >= (long)E) continue;                  // no barriers below: safe
        const int snd = senders[e];
        const int rcv = receivers[e];
        const float* ea = edge_attrs + (size_t)e * EDGE_DIM;
        const float ev0 = ea[8], ev1 = ea[9], ev2 = ea[10];
        const float* nf   = node_feats + (size_t)snd * NODE_DIM;
        float*       op   = out        + (size_t)rcv * OUT_DIM;
        const float* mixr = mixbuf + m * 256;

        #pragma unroll
        for (int ii = 0; ii < 2; ++ii) {
            const int i = lane + 32 * ii;
            const float s_i = nf[i];
            const float v0 = nf[64 + 3 * i];
            const float v1 = nf[64 + 3 * i + 1];
            const float v2 = nf[64 + 3 * i + 2];
            // scalar messages
            atomicAdd(op + i, s_i * mixr[i] * OUT_SCALE);
            const float tps = (v0 * ev0 + v1 * ev1 + v2 * ev2) * INV_SQRT3;
            atomicAdd(op + 64 + i, tps * mixr[64 + i] * OUT_SCALE);
            // vector messages: mv part
            const float mvw = mixr[128 + i] * OUT_SCALE;
            atomicAdd(op + 128 + 3 * i,     v0 * mvw);
            atomicAdd(op + 128 + 3 * i + 1, v1 * mvw);
            atomicAdd(op + 128 + 3 * i + 2, v2 * mvw);
            // vector messages: tp_v part (ms outer ev)
            const float tpw = s_i * mixr[192 + i] * OUT_SCALE;
            atomicAdd(op + 320 + 3 * i,     tpw * ev0);
            atomicAdd(op + 320 + 3 * i + 1, tpw * ev1);
            atomicAdd(op + 320 + 3 * i + 2, tpw * ev2);
        }
    }
}

__global__ void zero_out_kernel(float* __restrict__ p, long n) {
    long i = (long)blockIdx.x * blockDim.x + threadIdx.x;
    if (i < n) p[i] = 0.0f;
}

extern "C" void kernel_launch(void* const* d_in, const int* in_sizes, int n_in,
                              void* d_out, int out_size, void* d_ws, size_t ws_size,
                              hipStream_t stream) {
    (void)n_in; (void)d_ws; (void)ws_size;
    const float* node_feats = (const float*)d_in[0];
    const float* edge_attrs = (const float*)d_in[1];
    const int*   senders    = (const int*)d_in[2];
    const int*   receivers  = (const int*)d_in[3];
    const float* w0         = (const float*)d_in[4];
    const float* w1         = (const float*)d_in[5];
    const float* w2         = (const float*)d_in[6];
    const float* w3         = (const float*)d_in[7];
    float* out = (float*)d_out;

    const int  E     = in_sizes[2];
    const long n_out = (long)out_size;

    // out is accumulated with atomics -> zero it every call (graph-safe).
    zero_out_kernel<<<(int)((n_out + 255) / 256), 256, 0, stream>>>(out, n_out);

    const int tiles  = (E + 15) / 16;     // 16 edges per wave
    const int blocks = (tiles + 3) / 4;   // 4 waves (128 threads) per block
    mp_conv_kernel<<<blocks, 128, LDS_BYTES, stream>>>(
        node_feats, edge_attrs, senders, receivers, w0, w1, w2, w3, out, E);
}
